// MultiHeadAttention_29137058136064
// MI455X (gfx1250) — matrix operand order repaired and checked by reading, not compiled, not run on hardware
//
#include <hip/hip_runtime.h>

// ---------------------------------------------------------------------------
// MI455X (gfx1250, CDNA5) implementation of:
//   qkv = x @ w_qkv + b_qkv ; block-sparse flash attention ; out = o @ w_o + b_o
// B=2, S=2048, H=16, DK=64, D=1024, BS=128, Tr=Tc=16.
// Matmuls on v_wmma_f32_16x16x32_bf16.  Fully async-pipelined staging:
//   - direct tile copies: global_load_async_to_lds_b128 (ASYNCcnt)
//   - transposed fills:   inline-asm global_load_b128 -> regs (invisible to
//     the compiler's barrier waitcnt insertion) -> ds_store next iteration,
//     synchronized with explicit s_wait_loadcnt at the consumption point.
// ---------------------------------------------------------------------------

typedef __attribute__((ext_vector_type(16))) __bf16        v16bf;
typedef __attribute__((ext_vector_type(8)))  float         v8f;
typedef __attribute__((ext_vector_type(4)))  unsigned int  u32x4;

union Frag { v16bf v; u32x4 q[2]; };

#define DEV __device__ __forceinline__

static constexpr int NB   = 2;     // batch
static constexpr int SS   = 2048;  // seq
static constexpr int NH   = 16;    // heads
static constexpr int DK   = 64;    // head dim
static constexpr int DM   = 1024;  // model dim
static constexpr int BLK  = 128;   // flash block
static constexpr int TT   = 16;    // Tr = Tc

DEV unsigned short f2bf(float f) {
  unsigned u = __float_as_uint(f);
  unsigned r = u + 0x7FFFu + ((u >> 16) & 1u);   // round-to-nearest-even
  return (unsigned short)(r >> 16);
}

DEV v8f wmma_bf16(v16bf a, v16bf b, v8f c) {
  // (neg_a, A, neg_b, B, c_mod, C, reuse_a, reuse_b)
  return __builtin_amdgcn_wmma_f32_16x16x32_bf16(false, a, false, b, (short)0, c,
                                                 false, false);
}

DEV v8f zero8() {
  v8f r;
#pragma unroll
  for (int i = 0; i < 8; ++i) r[i] = 0.f;
  return r;
}

// Async global->LDS copy, 16B per lane (CDNA5 GLOBAL_LOAD_ASYNC_TO_LDS_B128,
// ASYNCcnt-tracked, bypasses VGPRs).  Completions are in-order, so a
// "wait <= N" drains everything except the last N per-wave copies.
DEV void async_copy_b128(void* lds_ptr, const void* gptr) {
  unsigned lds_addr = (unsigned)(unsigned long long)lds_ptr;
  asm volatile("global_load_async_to_lds_b128 %0, %1, off"
               :: "v"(lds_addr), "v"(gptr) : "memory");
}
template <int N>
DEV void wait_async_le() {
  asm volatile("s_wait_asynccnt %0" :: "i"(N) : "memory");
}

// Compiler-invisible 16B global load (LOADcnt).  The compiler's barrier
// handling cannot see it, so it is NOT force-drained at __syncthreads();
// the caller waits with wait_load0() before reading the result registers.
DEV u32x4 asm_gload_b128(const void* gptr) {
  u32x4 r;
  asm volatile("global_load_b128 %0, %1, off"
               : "=v"(r) : "v"(gptr) : "memory");
  return r;
}
DEV void wait_load0() {
  asm volatile("s_wait_loadcnt 0x0" ::: "memory");
}

// ---------------------------------------------------------------------------
// f32 -> bf16 conversion
// ---------------------------------------------------------------------------
__global__ void cvt_bf16_kernel(const float* __restrict__ src,
                                unsigned short* __restrict__ dst, int n) {
  int i = blockIdx.x * blockDim.x + threadIdx.x;
  if (i < n) dst[i] = f2bf(src[i]);
}

// ---------------------------------------------------------------------------
// Double-buffered WMMA GEMM main loop: C[128x128] += A[128xK] * B[Kx128]
// As/Bs each hold two 128-row buffers (rows [p*128, p*128+128)).
// K-loop runs two steps per trip with ping-pong register sets so the
// in-flight B registers are never copied before their wait.
// ---------------------------------------------------------------------------
DEV void gemm_tile(const unsigned short* __restrict__ A, int lda,
                   const unsigned short* __restrict__ Bm, int ldb,
                   int mBase, int nBase, int Kdim,
                   unsigned short (*As)[40], unsigned short (*Bs)[40],
                   v8f* acc) {
  const int tid  = threadIdx.x;
  const int lane = tid & 31;
  const int wid  = tid >> 5;
  const int hi   = (lane >> 4) & 1;
  const int lr   = lane & 15;
  const int m0   = wid * 16;
  const int niter = Kdim / 32;               // even (K multiple of 64)

  auto step = [&](int ic, u32x4* cur, u32x4* nxt) {
    const int p = ic & 1;
    wait_load0();                             // B(ic) regs have landed
#pragma unroll
    for (int i = 0; i < 2; ++i) {             // B(ic): regs -> LDS buf p
      int idx = tid + i * 256;
      int kk = idx >> 4, sg = idx & 15;
      const unsigned short* e = (const unsigned short*)&cur[i];
#pragma unroll
      for (int j = 0; j < 8; ++j) Bs[p * 128 + sg * 8 + j][kk] = e[j];
    }
    if (ic + 1 < niter) {                     // prefetch tile ic+1
      const int kb2 = (ic + 1) * 32;
#pragma unroll
      for (int i = 0; i < 2; ++i) {
        int idx = tid + i * 256;
        int kk = idx >> 4, sg = idx & 15;
        nxt[i] = asm_gload_b128(Bm + (size_t)(kb2 + kk) * ldb + nBase + sg * 8);
      }
#pragma unroll
      for (int i = 0; i < 2; ++i) {
        int idx = tid + i * 256;
        int r = idx >> 2, sg = idx & 3;
        async_copy_b128(&As[(1 - p) * 128 + r][sg * 8],
                        A + (size_t)(mBase + r) * lda + kb2 + sg * 8);
      }
      wait_async_le<2>();                     // A(ic) landed; A(ic+1) in flight
    } else {
      wait_async_le<0>();
    }
    __syncthreads();                          // dscnt-only wait here

    Frag fa;
    const unsigned short* ar = &As[p * 128 + m0 + lr][0];
    fa.q[0] = *(const u32x4*)(ar + 16 * hi);
    fa.q[1] = *(const u32x4*)(ar + 16 * hi + 8);
    Frag fb[8];
#pragma unroll
    for (int nt = 0; nt < 8; ++nt) {
      const unsigned short* br = &Bs[p * 128 + nt * 16 + lr][0];
      fb[nt].q[0] = *(const u32x4*)(br + 16 * hi);
      fb[nt].q[1] = *(const u32x4*)(br + 16 * hi + 8);
    }
#pragma unroll
    for (int nt = 0; nt < 8; ++nt)
      acc[nt] = wmma_bf16(fa.v, fb[nt].v, acc[nt]);
    __syncthreads();
  };

  u32x4 rbA[2], rbB[2];
  // ---- prologue: issue tile 0 ----
#pragma unroll
  for (int i = 0; i < 2; ++i) {               // B(0) -> regs (invisible load)
    int idx = tid + i * 256;
    int kk = idx >> 4, sg = idx & 15;
    rbA[i] = asm_gload_b128(Bm + (size_t)kk * ldb + nBase + sg * 8);
  }
#pragma unroll
  for (int i = 0; i < 2; ++i) {               // A(0) -> LDS buf0, async
    int idx = tid + i * 256;
    int r = idx >> 2, sg = idx & 3;
    async_copy_b128(&As[r][sg * 8], A + (size_t)(mBase + r) * lda + sg * 8);
  }

  for (int ic = 0; ic < niter; ic += 2) {     // ping-pong register sets
    step(ic, rbA, rbB);
    step(ic + 1, rbB, rbA);
  }
}

// ---------------------------------------------------------------------------
// Kernel 2: QKV projection + head scatter.  Q is pre-scaled by tau/sqrt(dk).
// Output layout Q/K/V: [B][H][S][DK] bf16.
// ---------------------------------------------------------------------------
__global__ __launch_bounds__(256)
void qkv_gemm_kernel(const unsigned short* __restrict__ xb,
                     const unsigned short* __restrict__ wb,
                     const float* __restrict__ bias,
                     const float* __restrict__ taup,
                     unsigned short* __restrict__ Qb,
                     unsigned short* __restrict__ Kb,
                     unsigned short* __restrict__ Vb) {
  __shared__ unsigned short As[256][40];   // 2 buffers x 128 rows
  __shared__ unsigned short Bs[256][40];
  v8f acc[8];
#pragma unroll
  for (int i = 0; i < 8; ++i) acc[i] = zero8();

  const int mBase = blockIdx.y * 128;
  const int nBase = blockIdx.x * 128;
  gemm_tile(xb, DM, wb, 3 * DM, mBase, nBase, DM, As, Bs, acc);

  const int lane = threadIdx.x & 31;
  const int wid  = threadIdx.x >> 5;
  const int hi   = (lane >> 4) & 1;
  const int lr   = lane & 15;
  const int m0   = wid * 16;
  const float qscale = taup[0] * 0.125f;   // tau / sqrt(64)

#pragma unroll
  for (int nt = 0; nt < 8; ++nt) {
    int col  = nBase + nt * 16 + lr;
    int part = col >> 10;                  // 0=q 1=k 2=v
    int c    = col & 1023;
    int h    = c >> 6;
    int dk   = c & 63;
    unsigned short* dst = (part == 0) ? Qb : ((part == 1) ? Kb : Vb);
    float mul = (part == 0) ? qscale : 1.f;
    float bv  = bias[col];
#pragma unroll
    for (int i = 0; i < 8; ++i) {
      int row = mBase + m0 + i + 8 * hi;   // C layout: M = vgpr + 8*laneHi
      int b = row >> 11, s = row & 2047;
      float v = (acc[nt][i] + bv) * mul;
      dst[(((size_t)(b * NH + h)) * SS + s) * DK + dk] = f2bf(v);
    }
  }
}

// ---------------------------------------------------------------------------
// Kernel 3: block-sparse flash attention, double-buffered K/V.
// Grid: (Tr, H, B), 256 threads (8 waves). LDS (dynamic, ~189KB of 320KB WGP):
//   Qs [128][72] | Ks 2x[128][72] | Vt 2x[64][136] | Sb [128][132] f32
//   Pb [128][136] | alpha[128] | rowL[128]
// ---------------------------------------------------------------------------
static constexpr int QS_OFF = 0;
static constexpr int KS_OFF = QS_OFF + 128 * 72 * 2;          // 18432
static constexpr int VT_OFF = KS_OFF + 2 * 128 * 72 * 2;      // 55296
static constexpr int SB_OFF = VT_OFF + 2 * 64 * 136 * 2;      // 90112
static constexpr int PB_OFF = SB_OFF + 128 * 132 * 4;         // 157696
static constexpr int AL_OFF = PB_OFF + 128 * 136 * 2;         // 192512
static constexpr int RL_OFF = AL_OFF + 128 * 4;               // 193024
static constexpr int SMEM_BYTES = RL_OFF + 128 * 4;           // 193536

DEV void k_issue_async(unsigned short* KsBuf, const unsigned short* Kbase,
                       int tid) {
#pragma unroll
  for (int i = 0; i < 4; ++i) {
    int idx = tid + i * 256;
    int r = idx >> 3, sg = idx & 7;
    async_copy_b128(&KsBuf[r * 72 + sg * 8], Kbase + (size_t)r * DK + sg * 8);
  }
}
DEV void v_issue_loads(const unsigned short* Vbase, u32x4* vld, int tid) {
#pragma unroll
  for (int i = 0; i < 4; ++i) {             // compiler-invisible loads
    int idx = tid + i * 256;
    int kk = idx >> 3, sg = idx & 7;
    vld[i] = asm_gload_b128(Vbase + (size_t)kk * DK + sg * 8);
  }
}
DEV void v_store_lds(unsigned short* VtBuf, const u32x4* vld, int tid) {
#pragma unroll
  for (int i = 0; i < 4; ++i) {
    int idx = tid + i * 256;
    int kk = idx >> 3, sg = idx & 7;
    const unsigned short* e = (const unsigned short*)&vld[i];
#pragma unroll
    for (int j = 0; j < 8; ++j) VtBuf[(sg * 8 + j) * 136 + kk] = e[j];
  }
}

__global__ __launch_bounds__(256)
void flash_attn_kernel(const unsigned short* __restrict__ Qb,
                       const unsigned short* __restrict__ Kb,
                       const unsigned short* __restrict__ Vb,
                       const int* __restrict__ mask,
                       unsigned short* __restrict__ Ob) {
  extern __shared__ char smem[];
  unsigned short* Qs = (unsigned short*)(smem + QS_OFF);
  unsigned short* Ks = (unsigned short*)(smem + KS_OFF);
  unsigned short* Vt = (unsigned short*)(smem + VT_OFF);
  float*          Sb = (float*)(smem + SB_OFF);
  unsigned short* Pb = (unsigned short*)(smem + PB_OFF);
  float*       alpha = (float*)(smem + AL_OFF);
  float*        rowL = (float*)(smem + RL_OFF);

  const int ir = blockIdx.x, h = blockIdx.y, b = blockIdx.z;
  const int tid  = threadIdx.x;
  const int lane = tid & 31;
  const int wid  = tid >> 5;
  const int hi   = (lane >> 4) & 1;
  const int lr   = lane & 15;
  const int m0   = wid * 16;
  const size_t headoff = ((size_t)(b * NH + h)) * SS * DK;

  // Q block resident for whole kernel (already pre-scaled by tau/sqrt(dk))
#pragma unroll
  for (int i = 0; i < 4; ++i) {
    int idx = tid + i * 256;
    int r = idx >> 3, sg = idx & 7;
    async_copy_b128(&Qs[r * 72 + sg * 8],
                    Qb + headoff + (size_t)(ir * BLK + r) * DK + sg * 8);
  }

  float mreg = -1e30f, lreg = 0.f;   // per-row softmax state (tid<128 owns row tid)
  v8f oacc[4];
#pragma unroll
  for (int i = 0; i < 4; ++i) oacc[i] = zero8();

  // Active column-block bitmask (uniform across block)
  unsigned am = 0;
  for (int j = 0; j < TT; ++j)
    if (mask[ir * TT + j] != 0) am |= 1u << j;

  u32x4 vcur[4], vnxt[4];
  int jc = -1;
  if (am) {                                 // prologue: issue first block
    jc = __builtin_ctz(am); am &= am - 1;
    k_issue_async(Ks, Kb + headoff + (size_t)jc * BLK * DK, tid);
    v_issue_loads(Vb + headoff + (size_t)jc * BLK * DK, vcur, tid);
  }

  int t = 0;
  while (jc >= 0) {
    const int p = t & 1;
    unsigned short* KsBuf = Ks + p * 128 * 72;
    unsigned short* VtBuf = Vt + p * 64 * 136;

    wait_load0();                           // V(t) regs landed (no-op if t>0)
    v_store_lds(VtBuf, vcur, tid);          // V(t): regs -> LDS

    int jn = -1;                            // prefetch next active block
    if (am) { jn = __builtin_ctz(am); am &= am - 1; }
    if (jn >= 0) {
      k_issue_async(Ks + (1 - p) * 128 * 72,
                    Kb + headoff + (size_t)jn * BLK * DK, tid);
      v_issue_loads(Vb + headoff + (size_t)jn * BLK * DK, vnxt, tid);
      wait_async_le<4>();                   // Q + K(t) landed; K(t+1) in flight
    } else {
      wait_async_le<0>();
    }
    __syncthreads();

    // S strip = Q(16x64) . K^T : 8 tiles x 2 k-steps per wave.
    Frag fa[2];
#pragma unroll
    for (int ksi = 0; ksi < 2; ++ksi) {
      const unsigned short* ar = Qs + (m0 + lr) * 72 + ksi * 32;
      fa[ksi].q[0] = *(const u32x4*)(ar + 16 * hi);
      fa[ksi].q[1] = *(const u32x4*)(ar + 16 * hi + 8);
    }
#pragma unroll
    for (int nt = 0; nt < 8; ++nt) {
      Frag fb[2];
#pragma unroll
      for (int ksi = 0; ksi < 2; ++ksi) {   // B[k][n] = K[n][k]
        const unsigned short* br = KsBuf + (nt * 16 + lr) * 72 + ksi * 32;
        fb[ksi].q[0] = *(const u32x4*)(br + 16 * hi);
        fb[ksi].q[1] = *(const u32x4*)(br + 16 * hi + 8);
      }
      v8f sacc = zero8();
      sacc = wmma_bf16(fa[0].v, fb[0].v, sacc);
      sacc = wmma_bf16(fa[1].v, fb[1].v, sacc);
#pragma unroll
      for (int i = 0; i < 8; ++i)
        Sb[(m0 + i + 8 * hi) * 132 + nt * 16 + lr] = sacc[i];
    }
    __syncthreads();

    // Online softmax, one thread per row
    if (tid < 128) {
      const float* srow = Sb + tid * 132;
      unsigned short* prow = Pb + tid * 136;
      float mij = -1e30f;
      for (int c = 0; c < 128; ++c) mij = fmaxf(mij, srow[c]);
      float mnew = fmaxf(mreg, mij);
      float a = __expf(mreg - mnew);
      float sum = 0.f;
      for (int c = 0; c < 128; ++c) {
        float pp = __expf(srow[c] - mnew);
        sum += pp;
        prow[c] = f2bf(pp);
      }
      lreg = a * lreg + sum;
      mreg = mnew;
      alpha[tid] = a;
    }
    __syncthreads();

    // O = O*alpha + P.V : 4 k-steps x 4 n-tiles per wave
    {
      float al[8];
#pragma unroll
      for (int i = 0; i < 8; ++i) al[i] = alpha[m0 + i + 8 * hi];
#pragma unroll
      for (int nt = 0; nt < 4; ++nt)
#pragma unroll
        for (int i = 0; i < 8; ++i) oacc[nt][i] *= al[i];

#pragma unroll
      for (int ks = 0; ks < 128; ks += 32) {
        Frag fa2;
        const unsigned short* pr = Pb + (m0 + lr) * 136;
        fa2.q[0] = *(const u32x4*)(pr + ks + 16 * hi);
        fa2.q[1] = *(const u32x4*)(pr + ks + 16 * hi + 8);
        Frag fb[4];
#pragma unroll
        for (int nt = 0; nt < 4; ++nt) {    // B[k][n] = Vt[n][k]
          const unsigned short* br = VtBuf + (nt * 16 + lr) * 136 + ks;
          fb[nt].q[0] = *(const u32x4*)(br + 16 * hi);
          fb[nt].q[1] = *(const u32x4*)(br + 16 * hi + 8);
        }
#pragma unroll
        for (int nt = 0; nt < 4; ++nt)
          oacc[nt] = wmma_bf16(fa2.v, fb[nt].v, oacc[nt]);
      }
    }
    __syncthreads();

    wait_load0();                           // V(t+1) regs landed before copy
#pragma unroll
    for (int i = 0; i < 4; ++i) vcur[i] = vnxt[i];
    jc = jn;
    ++t;
  }

  if (tid < 128) rowL[tid] = lreg;
  __syncthreads();

  float invl[8];
#pragma unroll
  for (int i = 0; i < 8; ++i) {
    float l = rowL[m0 + i + 8 * hi];
    invl[i] = (l > 0.f) ? (1.f / l) : 0.f;    // fully-masked rows -> 0
  }
#pragma unroll
  for (int nt = 0; nt < 4; ++nt) {
    int col = h * DK + nt * 16 + lr;          // [B][S][H*DK] layout
#pragma unroll
    for (int i = 0; i < 8; ++i) {
      int s = ir * BLK + m0 + i + 8 * hi;
      Ob[((size_t)b * SS + s) * DM + col] = f2bf(oacc[nt][i] * invl[i]);
    }
  }
}

// ---------------------------------------------------------------------------
// Kernel 4: output projection, f32 result
// ---------------------------------------------------------------------------
__global__ __launch_bounds__(256)
void oproj_kernel(const unsigned short* __restrict__ Ob,
                  const unsigned short* __restrict__ wob,
                  const float* __restrict__ bias,
                  float* __restrict__ out) {
  __shared__ unsigned short As[256][40];
  __shared__ unsigned short Bs[256][40];
  v8f acc[8];
#pragma unroll
  for (int i = 0; i < 8; ++i) acc[i] = zero8();

  const int mBase = blockIdx.y * 128;
  const int nBase = blockIdx.x * 128;
  gemm_tile(Ob, DM, wob, DM, mBase, nBase, DM, As, Bs, acc);

  const int lane = threadIdx.x & 31;
  const int wid  = threadIdx.x >> 5;
  const int hi   = (lane >> 4) & 1;
  const int lr   = lane & 15;
  const int m0   = wid * 16;

#pragma unroll
  for (int nt = 0; nt < 8; ++nt) {
    int col = nBase + nt * 16 + lr;
    float bv = bias[col];
#pragma unroll
    for (int i = 0; i < 8; ++i) {
      int row = mBase + m0 + i + 8 * hi;
      out[(size_t)row * DM + col] = acc[nt][i] + bv;
    }
  }
}

// ---------------------------------------------------------------------------
// Host launcher
// ---------------------------------------------------------------------------
extern "C" void kernel_launch(void* const* d_in, const int* in_sizes, int n_in,
                              void* d_out, int out_size, void* d_ws, size_t ws_size,
                              hipStream_t stream) {
  const float* x     = (const float*)d_in[0];
  const float* w_qkv = (const float*)d_in[1];
  const float* b_qkv = (const float*)d_in[2];
  const float* w_o   = (const float*)d_in[3];
  const float* b_o   = (const float*)d_in[4];
  const float* tau   = (const float*)d_in[5];
  const int*   bmask = (const int*)d_in[6];
  float*       out   = (float*)d_out;

  char* ws = (char*)d_ws;
  const size_t nX    = (size_t)NB * SS * DM;       // 4M
  const size_t nWqkv = (size_t)DM * 3 * DM;        // 3M
  const size_t nWo   = (size_t)DM * DM;            // 1M
  const size_t nHead = (size_t)NB * NH * SS * DK;  // 4M

  unsigned short* xb    = (unsigned short*)(ws);
  unsigned short* wqkvb = xb + nX;
  unsigned short* wob   = wqkvb + nWqkv;
  unsigned short* Qb    = wob + nWo;
  unsigned short* Kb    = Qb + nHead;
  unsigned short* Vb    = Kb + nHead;
  unsigned short* Ob    = Vb + nHead;
  (void)ws_size; (void)n_in; (void)in_sizes; (void)out_size;

  // 1) f32 -> bf16
  cvt_bf16_kernel<<<(int)((nX + 255) / 256), 256, 0, stream>>>(x, xb, (int)nX);
  cvt_bf16_kernel<<<(int)((nWqkv + 255) / 256), 256, 0, stream>>>(w_qkv, wqkvb, (int)nWqkv);
  cvt_bf16_kernel<<<(int)((nWo + 255) / 256), 256, 0, stream>>>(w_o, wob, (int)nWo);

  // 2) QKV GEMM (M=4096, N=3072, K=1024)
  qkv_gemm_kernel<<<dim3(3 * DM / 128, NB * SS / 128), 256, 0, stream>>>(
      xb, wqkvb, b_qkv, tau, Qb, Kb, Vb);

  // 3) block-sparse flash attention (needs >64KB dynamic LDS; WGP has 320KB)
  static int attn_smem_set = 0;
  if (!attn_smem_set) {
    (void)hipFuncSetAttribute((const void*)flash_attn_kernel,
                              hipFuncAttributeMaxDynamicSharedMemorySize,
                              SMEM_BYTES);
    attn_smem_set = 1;
  }
  flash_attn_kernel<<<dim3(TT, NH, NB), 256, SMEM_BYTES, stream>>>(
      Qb, Kb, Vb, bmask, Ob);

  // 4) output projection (M=4096, N=1024, K=1024)
  oproj_kernel<<<dim3(DM / 128, NB * SS / 128), 256, 0, stream>>>(
      Ob, wob, b_o, out);
}
